// ScaledDotProductAttention_32684701122973
// MI455X (gfx1250) — matrix operand order repaired and checked by reading, not compiled, not run on hardware
//
#include <hip/hip_runtime.h>

typedef __attribute__((ext_vector_type(16))) __bf16 v16bf;
typedef __attribute__((ext_vector_type(8)))  float  v8f;
typedef __attribute__((ext_vector_type(8)))  int    v8i;

#define SEQ    2048
#define HEADS  16
#define DH     64
#define QTILE  128     // Q rows per block (8 waves x 16 rows)
#define KT     64      // keys per KV iteration
#define NEGBIG (-3.0e38f)

__global__ __launch_bounds__(256)
void fa_fwd_bf16_wmma(const float* __restrict__ K, const float* __restrict__ Q,
                      const float* __restrict__ V, const float* __restrict__ M,
                      float* __restrict__ O)
{
    __shared__ __bf16 sKb[KT * DH];          // 8 KB: K tile, row-major (key, d), bf16
    __shared__ __bf16 sVt[DH * KT];          // 8 KB: V tile, TRANSPOSED (d, key), bf16
    __shared__ __bf16 sP[8][16 * KT];        // 16 KB: per-wave P scratch (C->A relayout)

    const int tid  = threadIdx.x;
    const int lane = tid & 31;
    const int wv   = tid >> 5;
    const int n    = lane & 15;              // column / row-in-tile index
    const int g    = lane >> 4;              // lane-half select

    const int nQblk = SEQ / QTILE;
    const int bh    = blockIdx.x / nQblk;    // fused (b,h)
    const int qblk  = blockIdx.x % nQblk;
    const int b     = bh / HEADS;

    const float* gK = K + (size_t)bh * SEQ * DH;
    const float* gQ = Q + (size_t)bh * SEQ * DH;
    const float* gV = V + (size_t)bh * SEQ * DH;
    const float* gM = M + (size_t)b  * SEQ;
    float*       gO = O + (size_t)bh * SEQ * DH;

    const int qbase = qblk * QTILE + wv * 16;

    // ---- Q A-fragments: 16x64 bf16 as two 16x32 frags (ISA 16-bit A layout) ----
    v16bf qa0, qa1;
    {
        const float* qrow = gQ + (size_t)(qbase + n) * DH;   // lane -> row M = n
        #pragma unroll
        for (int j = 0; j < 8; ++j) {
            int kk = (j < 4) ? (2*j + 8*g) : (16 + 2*(j-4) + 8*g);
            float2 x0 = *(const float2*)(qrow + kk);
            float2 x1 = *(const float2*)(qrow + 32 + kk);
            qa0[2*j] = (__bf16)x0.x; qa0[2*j+1] = (__bf16)x0.y;
            qa1[2*j] = (__bf16)x1.x; qa1[2*j+1] = (__bf16)x1.y;
        }
    }

    // Constant all-ones B fragment: rowsum(P) = P x ones via WMMA
    v16bf ones;
    #pragma unroll
    for (int i = 0; i < 16; ++i) ones[i] = (__bf16)1.0f;

    v8f acc[4];
    #pragma unroll
    for (int t = 0; t < 4; ++t) acc[t] = (v8f){};
    float mrow[8], lrow[8];
    #pragma unroll
    for (int r = 0; r < 8; ++r) { mrow[r] = NEGBIG; lrow[r] = 0.0f; }

    const int nIter = SEQ / KT;
    for (int t = 0; t < nIter; ++t) {
        const int kv0 = t * KT;

        // ---- stage K/V tiles into LDS as bf16 (convert ONCE per block) ----
        {
            const float4* srcK = (const float4*)(gK + (size_t)kv0 * DH);
            const float4* srcV = (const float4*)(gV + (size_t)kv0 * DH);
            #pragma unroll
            for (int p = 0; p < (KT * DH / 4) / 256; ++p) {
                const int i = tid + p * 256;          // float4 index, elems 4i..4i+3
                float4 kf = srcK[i];
                __bf16 kb4[4] = { (__bf16)kf.x, (__bf16)kf.y,
                                  (__bf16)kf.z, (__bf16)kf.w };
                *(uint2*)&sKb[4 * i] = *(const uint2*)kb4;
                float4 vf = srcV[i];
                const int key = (4 * i) / DH;
                const int d0  = (4 * i) % DH;
                sVt[(d0 + 0) * KT + key] = (__bf16)vf.x;
                sVt[(d0 + 1) * KT + key] = (__bf16)vf.y;
                sVt[(d0 + 2) * KT + key] = (__bf16)vf.z;
                sVt[(d0 + 3) * KT + key] = (__bf16)vf.w;
            }
        }
        if (t + 1 < nIter) {
            __builtin_prefetch(gK + (size_t)(kv0 + KT) * DH + tid * 64, 0, 1);
            __builtin_prefetch(gV + (size_t)(kv0 + KT) * DH + tid * 64, 0, 1);
        }
        __syncthreads();

        // ---- batch-load ALL 8 K B-fragments, then 8 back-to-back QK WMMAs ----
        v16bf kb[8];
        #pragma unroll
        for (int dstep = 0; dstep < 2; ++dstep) {
            #pragma unroll
            for (int nt = 0; nt < 4; ++nt) {
                v8i tb;
                #pragma unroll
                for (int j = 0; j < 8; ++j) {
                    int dd = dstep * 32 + ((j < 4) ? (2*j + 8*g) : (16 + 2*(j-4) + 8*g));   // reduction dim (d)
                    tb[j] = *(const int*)&sKb[(size_t)(nt * 16 + n) * DH + dd];
                }
                kb[dstep * 4 + nt] = __builtin_bit_cast(v16bf, tb);
            }
        }
        v8f c[4];
        #pragma unroll
        for (int nt = 0; nt < 4; ++nt) c[nt] = (v8f){};
        #pragma unroll
        for (int dstep = 0; dstep < 2; ++dstep) {
            const v16bf a = dstep ? qa1 : qa0;
            #pragma unroll
            for (int nt = 0; nt < 4; ++nt)
                c[nt] = __builtin_amdgcn_wmma_f32_16x16x32_bf16(false, a, false,
                                                                kb[dstep * 4 + nt],
                                                                (short)0, c[nt],
                                                                false, false);
        }

        // ---- batch-load ALL 8 V B-fragments now; latency hides under softmax ----
        v16bf vb[8];
        #pragma unroll
        for (int ks = 0; ks < 2; ++ks) {
            #pragma unroll
            for (int dt = 0; dt < 4; ++dt) {
                v8i tv;
                #pragma unroll
                for (int j = 0; j < 8; ++j)
                    tv[j] = *(const int*)&sVt[(size_t)(dt * 16 + n) * KT
                                              + ks * 32 + ((j < 4) ? (2*j + 8*g) : (16 + 2*(j-4) + 8*g))];
                vb[ks * 4 + dt] = __builtin_bit_cast(v16bf, tv);
            }
        }

        // ---- online softmax: row max via butterfly, exp, scatter P to LDS ----
        float add[4], alpha[8];
        #pragma unroll
        for (int nt = 0; nt < 4; ++nt)
            add[nt] = (1.0f - gM[kv0 + nt * 16 + n]) * NEGBIG;
        #pragma unroll
        for (int r = 0; r < 8; ++r) {
            float s0 = c[0][r] * 0.125f + add[0];    // 1/sqrt(64)
            float s1 = c[1][r] * 0.125f + add[1];
            float s2 = c[2][r] * 0.125f + add[2];
            float s3 = c[3][r] * 0.125f + add[3];
            float rm = fmaxf(fmaxf(s0, s1), fmaxf(s2, s3));
            rm = fmaxf(rm, __shfl_xor(rm, 1, 32));
            rm = fmaxf(rm, __shfl_xor(rm, 2, 32));
            rm = fmaxf(rm, __shfl_xor(rm, 4, 32));
            rm = fmaxf(rm, __shfl_xor(rm, 8, 32));
            float mn = fmaxf(mrow[r], rm);
            alpha[r] = __expf(mrow[r] - mn);
            mrow[r]  = mn;
            float e0 = __expf(s0 - mn);
            float e1 = __expf(s1 - mn);
            float e2 = __expf(s2 - mn);
            float e3 = __expf(s3 - mn);
            // scatter P (row-major 16x64 bf16) for C->A relayout
            __bf16* pr = &sP[wv][(r + 8 * g) * KT];
            pr[n]      = (__bf16)e0;
            pr[16 + n] = (__bf16)e1;
            pr[32 + n] = (__bf16)e2;
            pr[48 + n] = (__bf16)e3;
            acc[0][r] *= alpha[r]; acc[1][r] *= alpha[r];
            acc[2][r] *= alpha[r]; acc[3][r] *= alpha[r];
        }
        asm volatile("s_wait_dscnt 0" ::: "memory");   // P store -> P load (same wave)

        // ---- P A-fragments (two 16x32 frags over 64 keys) ----
        v16bf pa0, pa1;
        {
            const __bf16* pw = sP[wv] + (size_t)n * KT;   // lane -> row M = n
            v8i tp0, tp1;
            #pragma unroll
            for (int j = 0; j < 8; ++j) {
                int kk = (j < 4) ? (2*j + 8*g) : (16 + 2*(j-4) + 8*g);
                tp0[j] = *(const int*)&pw[kk];
                tp1[j] = *(const int*)&pw[32 + kk];
            }
            pa0 = __builtin_bit_cast(v16bf, tp0);
            pa1 = __builtin_bit_cast(v16bf, tp1);
        }

        // ---- row sums via WMMA: csum = P x ones (replaces sum butterflies) ----
        v8f csum = (v8f){};
        csum = __builtin_amdgcn_wmma_f32_16x16x32_bf16(false, pa0, false, ones,
                                                       (short)0, csum, false, false);
        csum = __builtin_amdgcn_wmma_f32_16x16x32_bf16(false, pa1, false, ones,
                                                       (short)0, csum, false, false);
        #pragma unroll
        for (int r = 0; r < 8; ++r)
            lrow[r] = lrow[r] * alpha[r] + csum[r];

        // ---- O += P V : 8 back-to-back bf16 WMMAs over K=64 keys ----
        #pragma unroll
        for (int ks = 0; ks < 2; ++ks) {
            const v16bf pa = ks ? pa1 : pa0;
            #pragma unroll
            for (int dt = 0; dt < 4; ++dt)
                acc[dt] = __builtin_amdgcn_wmma_f32_16x16x32_bf16(false, pa, false,
                                                                  vb[ks * 4 + dt],
                                                                  (short)0, acc[dt],
                                                                  false, false);
        }
        __syncthreads();   // protect sKb/sVt before next stage
    }

    // ---- epilogue: O = acc / l ----
    #pragma unroll
    for (int r = 0; r < 8; ++r) {
        float inv = 1.0f / lrow[r];
        float* orow = gO + (size_t)(qbase + r + 8 * g) * DH;
        orow[ 0 + n] = acc[0][r] * inv;
        orow[16 + n] = acc[1][r] * inv;
        orow[32 + n] = acc[2][r] * inv;
        orow[48 + n] = acc[3][r] * inv;
    }
}

extern "C" void kernel_launch(void* const* d_in, const int* in_sizes, int n_in,
                              void* d_out, int out_size, void* d_ws, size_t ws_size,
                              hipStream_t stream) {
    // setup_inputs order: k, q, v, attn_mask (all fp32)
    const float* k = (const float*)d_in[0];
    const float* q = (const float*)d_in[1];
    const float* v = (const float*)d_in[2];
    const float* m = (const float*)d_in[3];
    float*       o = (float*)d_out;

    const int B = 2, H = HEADS;
    dim3 grid(B * H * (SEQ / QTILE));
    dim3 block(256);
    fa_fwd_bf16_wmma<<<grid, block, 0, stream>>>(k, q, v, m, o);
}
